// GroupedQueryAttention_88304527605957
// MI455X (gfx1250) — compile-verified, matrix operand order checked by reading, not run on hardware
//
#include <hip/hip_runtime.h>

typedef _Float16 v8h  __attribute__((ext_vector_type(8)));
typedef _Float16 v16h __attribute__((ext_vector_type(16)));
typedef float    v8f  __attribute__((ext_vector_type(8)));

#define D_MODEL   1024
#define NUM_HEADS 16
#define N_REP     4
#define KV_HEADS  4
#define HEAD_DIM  64
#define KV_DIM    256
#define BATCH     2
#define SEQ       2048
#define MTOT      (BATCH * SEQ)

// ---------------------------------------------------------------------------
// CDNA5 async global->LDS staging via inline asm (ASYNCcnt-tracked).
// GV mode: LDS byte address in VDST VGPR (low 32 bits of flat shared ptr),
// 64-bit global address in VADDR pair; INST_OFFSET added to both sides.
// ---------------------------------------------------------------------------
static __device__ inline void lds_copy16h(const _Float16* g, _Float16* l) {
#if defined(__gfx1250__)
    asm volatile(
        "global_load_async_to_lds_b128 %0, %1, off\n\t"
        "global_load_async_to_lds_b128 %0, %1, off offset:16"
        :
        : "v"((unsigned)(unsigned long long)l),
          "v"((unsigned long long)g)
        : "memory");
#else
    *(v8h*)l = *(const v8h*)g;
    *(v8h*)(l + 8) = *(const v8h*)(g + 8);
#endif
}

static __device__ inline void async_wait_all() {
#if defined(__gfx1250__)
    asm volatile("s_wait_asynccnt 0x0" ::: "memory");
#endif
}

// Build a 16-half WMMA fragment from two contiguous 16-byte chunks.
// CDNA5 16-bit A/B layout: lane L holds row L%16, K-subset
// {h*8..h*8+7} u {16+h*8..16+h*8+7}, h = L/16 (ISA 7.12.2).
static __device__ inline v16h make_frag(const _Float16* lo, const _Float16* hi) {
    v8h a = *(const v8h*)lo;
    v8h b = *(const v8h*)hi;
    v16h r;
#pragma unroll
    for (int i = 0; i < 8; ++i) { r[i] = a[i]; r[i + 8] = b[i]; }
    return r;
}

static __device__ inline v8f wmma_f16(v16h a, v16h b, v8f c) {
    return __builtin_amdgcn_wmma_f32_16x16x32_f16(false, a, false, b,
                                                  (short)0, c, false, false);
}

// Convert+store 16 A elements (f32 source) as two b128 LDS stores.
static __device__ inline void stage16(const float* src, _Float16* dst) {
    v8h h0, h1;
#pragma unroll
    for (int j = 0; j < 8; ++j) { h0[j] = (_Float16)src[j]; h1[j] = (_Float16)src[j + 8]; }
    *(v8h*)dst = h0;
    *(v8h*)(dst + 8) = h1;
}
// f16 source: pure copy -> async-to-LDS path.
static __device__ inline void stage16(const _Float16* src, _Float16* dst) {
    lds_copy16h(src, dst);
}

// ---------------------------------------------------------------------------
// Tiled GEMM: C[M,N] = (A[M,K] @ W[K,N] + bias) * out_scale
// Block = 256 threads (8 waves). Tile 128x128, K-chunk 32.
// Wave w owns a 32x64 sub-tile: 2 A-frags x 4 B-frags -> 8 WMMA / K-chunk.
// W tile loaded column-wise from global (coalesced across lanes), stored
// contiguously to LDS -> no LDS scatter.
// MODE (compile-time): 0 = f32 row-major, 1 = f16 row-major,
//                      2 = f16 transposed ([N][M], for V^T).
// ---------------------------------------------------------------------------
template <typename TA, int MODE>
__global__ __launch_bounds__(256)
void gemm_wmma(const TA* __restrict__ A, const float* __restrict__ W,
               const float* __restrict__ bias, void* __restrict__ Cout,
               int M, int N, int K, float out_scale) {
    __shared__ _Float16 Ah[128 * 32];   // [row][k]
    __shared__ _Float16 Bt[128 * 32];   // [n][k]

    const int tid  = threadIdx.x;
    const int lane = tid & 31;
    const int wave = tid >> 5;
    const int half = lane >> 4;
    const int mrow = lane & 15;
    const int bm = blockIdx.y * 128;
    const int bn = blockIdx.x * 128;
    const int mstrip = (wave & 3) * 32;
    const int nhalf  = (wave >> 2) * 64;

    v8f acc[2][4] = {};

    const int arow = tid >> 1, acb = (tid & 1) * 16;       // A: 128x32, 16/thread
    const int wn = tid & 127, wkseg = (tid >> 7) * 16;     // W: 32x128 col-wise

    const TA*    asrc = A + (size_t)(bm + arow) * K + acb;
    const float* wsrc = W + (size_t)wkseg * N + bn + wn;

    for (int k0 = 0; k0 < K; k0 += 32) {
        __syncthreads();
        // ---- A tile -> f16 LDS (two b128 stores, or async copy for f16 A)
        stage16(asrc + k0, &Ah[arow * 32 + acb]);
        // ---- W tile: column gather (coalesced), contiguous LDS store
        {
            const float* wp = wsrc + (size_t)k0 * N;
            float tmp[16];
#pragma unroll
            for (int j = 0; j < 16; ++j) { tmp[j] = *wp; wp += N; }
            stage16(tmp, &Bt[wn * 32 + wkseg]);
        }
        if (k0 + 32 < K) {
            __builtin_prefetch(asrc + k0 + 32, 0, 0);
            __builtin_prefetch(wsrc + (size_t)(k0 + 32) * N, 0, 0);
        }
        async_wait_all();
        __syncthreads();

        v16h aA[2];
#pragma unroll
        for (int i = 0; i < 2; ++i) {
            const _Float16* ap = &Ah[(mstrip + i * 16 + mrow) * 32 + half * 8];
            aA[i] = make_frag(ap, ap + 16);
        }
#pragma unroll
        for (int n0 = 0; n0 < 4; ++n0) {
            const _Float16* bp = &Bt[(nhalf + n0 * 16 + mrow) * 32 + half * 8];
            v16h bB = make_frag(bp, bp + 16);
#pragma unroll
            for (int i = 0; i < 2; ++i)
                acc[i][n0] = wmma_f16(aA[i], bB, acc[i][n0]);
        }
    }

    // ---- epilogue: C layout VGPR r -> (M = r + half*8, N = lane%16)
#pragma unroll
    for (int i = 0; i < 2; ++i) {
        const int row0 = bm + mstrip + i * 16 + half * 8;
#pragma unroll
        for (int n0 = 0; n0 < 4; ++n0) {
            const int col = bn + nhalf + n0 * 16 + mrow;
            const float bv = bias[col];
            if constexpr (MODE == 0) {
                float* p = (float*)Cout + (size_t)row0 * N + col;
#pragma unroll
                for (int r = 0; r < 8; ++r) {
                    *p = (acc[i][n0][r] + bv) * out_scale;
                    p += N;
                }
            } else if constexpr (MODE == 1) {
                _Float16* p = (_Float16*)Cout + (size_t)row0 * N + col;
#pragma unroll
                for (int r = 0; r < 8; ++r) {
                    *p = (_Float16)((acc[i][n0][r] + bv) * out_scale);
                    p += N;
                }
            } else {
                // transposed: 8 consecutive rows -> one b128 store
                _Float16* p = (_Float16*)Cout + (size_t)col * M + row0;
                v8h hv;
#pragma unroll
                for (int r = 0; r < 8; ++r)
                    hv[r] = (_Float16)((acc[i][n0][r] + bv) * out_scale);
                *(v8h*)p = hv;
            }
        }
    }
}

// ---------------------------------------------------------------------------
// Flash GQA: block = 128 threads (4 waves); wave w owns query tile
// q0 + w*16 of one (batch, head). All 4 waves share one staged 32-key
// K/V block (cuts L2 re-reads 4x). V arrives pre-transposed in global
// ([dim][seq]) so staging is pure vectorized async copies.
// Scale 1/sqrt(d_model) pre-folded into Qh.
// ---------------------------------------------------------------------------
__global__ __launch_bounds__(128)
void gqa_flash(const _Float16* __restrict__ Qh, const _Float16* __restrict__ Kh,
               const _Float16* __restrict__ Vt, _Float16* __restrict__ Ctx) {
    __shared__ _Float16 Ks[32 * 64];     // [key][dim]
    __shared__ _Float16 Vs[64 * 32];     // [dim][key]
    __shared__ _Float16 Pt[4][16 * 32];  // per-wave [m][key]

    const int tid  = threadIdx.x;
    const int lane = tid & 31;
    const int wave = tid >> 5;
    const int half = lane >> 4;
    const int mrow = lane & 15;
    const int q0 = blockIdx.x * 64 + wave * 16;
    const int h  = blockIdx.y;
    const int b  = blockIdx.z;
    const int g  = h / N_REP;

    // Q fragments: 2 K-chunks of 32 dims (head_dim = 64)
    const size_t qbase = (size_t)(b * SEQ + q0 + mrow) * D_MODEL + h * HEAD_DIM;
    v16h aQ[2];
#pragma unroll
    for (int c = 0; c < 2; ++c) {
        const _Float16* p = Qh + qbase + c * 32 + half * 8;
        aQ[c] = make_frag(p, p + 16);
    }

    v8f accO[4] = {};
    float m_run[8], l_run[8];
#pragma unroll
    for (int r = 0; r < 8; ++r) { m_run[r] = -3.0e38f; l_run[r] = 0.0f; }

    // staging split: K tile 32x64 (krow/kcb), V tile 64x32 (drow/dcb)
    const int krow = tid >> 2, kcb = (tid & 3) * 16;
    const int drow = tid >> 1, dcb = (tid & 1) * 16;
    const _Float16* ksrc = Kh + (size_t)(b * SEQ + krow) * KV_DIM + g * HEAD_DIM + kcb;
    const _Float16* vsrc = Vt + (size_t)(g * HEAD_DIM + drow) * MTOT + b * SEQ + dcb;

    for (int kb = 0; kb < SEQ / 32; ++kb) {
        const int key0 = kb * 32;
        __syncthreads();
        lds_copy16h(ksrc + (size_t)key0 * KV_DIM, &Ks[krow * 64 + kcb]);
        lds_copy16h(vsrc + key0, &Vs[drow * 32 + dcb]);
        async_wait_all();
        __syncthreads();

        // ---- scores: S(16x32) = Q(16x64) @ K^T
        v8f s0 = {}, s1 = {};
#pragma unroll
        for (int c = 0; c < 2; ++c) {
            const _Float16* k0p = &Ks[(mrow) * 64 + c * 32 + half * 8];
            s0 = wmma_f16(aQ[c], make_frag(k0p, k0p + 16), s0);
            const _Float16* k1p = &Ks[(16 + mrow) * 64 + c * 32 + half * 8];
            s1 = wmma_f16(aQ[c], make_frag(k1p, k1p + 16), s1);
        }

        // ---- online softmax (rows: lanes 0-15 -> M=r, 16-31 -> M=r+8)
#pragma unroll
        for (int r = 0; r < 8; ++r) {
            float mx = fmaxf(s0[r], s1[r]);
#pragma unroll
            for (int off = 8; off >= 1; off >>= 1)
                mx = fmaxf(mx, __shfl_xor(mx, off, 32));
            const float mnew = fmaxf(m_run[r], mx);
            const float corr = __expf(m_run[r] - mnew);
            const float p0 = __expf(s0[r] - mnew);
            const float p1 = __expf(s1[r] - mnew);
            float sum = p0 + p1;
#pragma unroll
            for (int off = 8; off >= 1; off >>= 1)
                sum += __shfl_xor(sum, off, 32);
            l_run[r] = l_run[r] * corr + sum;
            m_run[r] = mnew;
#pragma unroll
            for (int j = 0; j < 4; ++j) accO[j][r] *= corr;
            const int m = half * 8 + r;
            Pt[wave][m * 32 + mrow]      = (_Float16)p0;
            Pt[wave][m * 32 + 16 + mrow] = (_Float16)p1;
        }

        // ---- ctx: O(16x64) += P(16x32) @ V(32x64)
        const _Float16* pp = &Pt[wave][mrow * 32 + half * 8];
        v16h aP = make_frag(pp, pp + 16);
#pragma unroll
        for (int j = 0; j < 4; ++j) {
            const _Float16* vp = &Vs[(j * 16 + mrow) * 32 + half * 8];
            accO[j] = wmma_f16(aP, make_frag(vp, vp + 16), accO[j]);
        }
    }

    // ---- normalize + store ctx (f16, row-major)
#pragma unroll
    for (int r = 0; r < 8; ++r) {
        const float inv = 1.0f / l_run[r];
        _Float16* p = Ctx + (size_t)(b * SEQ + q0 + half * 8 + r) * D_MODEL
                          + h * HEAD_DIM + mrow;
#pragma unroll
        for (int j = 0; j < 4; ++j)
            p[j * 16] = (_Float16)(accO[j][r] * inv);
    }
}

// ---------------------------------------------------------------------------
extern "C" void kernel_launch(void* const* d_in, const int* in_sizes, int n_in,
                              void* d_out, int out_size, void* d_ws, size_t ws_size,
                              hipStream_t stream) {
    (void)in_sizes; (void)n_in; (void)out_size; (void)ws_size;
    const float* x  = (const float*)d_in[0];
    const float* Wq = (const float*)d_in[1];
    const float* bq = (const float*)d_in[2];
    const float* Wk = (const float*)d_in[3];
    const float* bk = (const float*)d_in[4];
    const float* Wv = (const float*)d_in[5];
    const float* bv = (const float*)d_in[6];
    const float* Wo = (const float*)d_in[7];
    const float* bo = (const float*)d_in[8];
    float* out = (float*)d_out;

    // Workspace (f16): Q | K | V^T | Ctx  (~20.9 MB)
    _Float16* Qh  = (_Float16*)d_ws;
    _Float16* Kh  = Qh + (size_t)MTOT * D_MODEL;
    _Float16* Vth = Kh + (size_t)MTOT * KV_DIM;   // [KV_DIM][MTOT]
    _Float16* Ctx = Vth + (size_t)MTOT * KV_DIM;

    const dim3 blk(256);
    const float qscale = 0.03125f;  // 1/sqrt(d_model) = 1/32, folded into Q

    gemm_wmma<float, 1><<<dim3(D_MODEL / 128, MTOT / 128), blk, 0, stream>>>(
        x, Wq, bq, Qh, MTOT, D_MODEL, D_MODEL, qscale);
    gemm_wmma<float, 1><<<dim3(KV_DIM / 128, MTOT / 128), blk, 0, stream>>>(
        x, Wk, bk, Kh, MTOT, KV_DIM, D_MODEL, 1.0f);
    gemm_wmma<float, 2><<<dim3(KV_DIM / 128, MTOT / 128), blk, 0, stream>>>(
        x, Wv, bv, Vth, MTOT, KV_DIM, D_MODEL, 1.0f);

    gqa_flash<<<dim3(SEQ / 64, NUM_HEADS, BATCH), dim3(128), 0, stream>>>(
        Qh, Kh, Vth, Ctx);

    gemm_wmma<_Float16, 0><<<dim3(D_MODEL / 128, MTOT / 128), blk, 0, stream>>>(
        Ctx, Wo, bo, out, MTOT, D_MODEL, D_MODEL, 1.0f);
}
